// XORSignatures_51934744543460
// MI455X (gfx1250) — 8-bit matrix operand order repaired and checked by reading, compiled, not run on hardware
//
#include <hip/hip_runtime.h>

#define N_TOK   131072
#define DIM_SZ  1024
#define T_TILES 256
#define K_TOT   2048

typedef int   v16i __attribute__((ext_vector_type(16)));
typedef float v8f  __attribute__((ext_vector_type(8)));

#define FP8_P1 0x38u
#define FP8_M1 0xB8u

__global__ __launch_bounds__(256) void build_sigs_kernel(
    const float* __restrict__ base,
    const float* __restrict__ deltas,
    unsigned char* __restrict__ sigB)
{
    int gid = blockIdx.x * blockDim.x + threadIdx.x;
    int t   = gid >> 10;
    int dim = gid & (DIM_SZ - 1);

    float b  = base[dim];
    float bs = (float)((b > 0.0f) - (b < 0.0f));
    float v  = bs;
    if (t > 0) {
        float dl  = deltas[(size_t)(t - 1) * DIM_SZ + dim];
        float dsg = (float)((dl > 0.0f) - (dl < 0.0f));
        float ad  = fabsf(dsg);
        v = bs * (1.0f - ad) + dsg * ad;
    }
    sigB[(size_t)t * K_TOT + dim]          = (v > 0.0f) ? (unsigned char)FP8_M1 : (unsigned char)FP8_P1;
    sigB[(size_t)t * K_TOT + DIM_SZ + dim] = (v < 0.0f) ? (unsigned char)FP8_M1 : (unsigned char)FP8_P1;
}

#define KSTEP      128
#define LDS_STRIDE 144

__global__ __launch_bounds__(256) void hamming_gemm_kernel(
    const float* __restrict__ x,
    const unsigned char* __restrict__ sigB,
    int* __restrict__ idx_out,
    float* __restrict__ dist_out)
{
    __shared__ __align__(16) unsigned char ldsB[T_TILES * LDS_STRIDE];

    const int tid  = threadIdx.x;
    const int wave = tid >> 5;
    const int lane = tid & 31;
    const int half = lane >> 4;
    const int l16  = lane & 15;
    const int rowstrip = blockIdx.x * 128 + wave * 16;

    const float* xrow = x + (size_t)(rowstrip + l16) * DIM_SZ;

    v8f acc[16];
#pragma unroll
    for (int c = 0; c < 16; ++c)
#pragma unroll
        for (int r = 0; r < 8; ++r) acc[c][r] = 0.0f;

    for (int ks = 0; ks < (K_TOT / KSTEP); ++ks) {
        const int kbase = ks * KSTEP;

        __syncthreads();
        {
            const int4* src = (const int4*)(sigB + (size_t)tid * K_TOT + kbase);
            int4* dst = (int4*)(ldsB + tid * LDS_STRIDE);
#pragma unroll
            for (int q = 0; q < 8; ++q) dst[q] = src[q];
        }
        __syncthreads();

        const int   plane = (kbase >= DIM_SZ) ? 1 : 0;
        const int   dbase = kbase - plane * DIM_SZ;
        const float psgn  = plane ? -1.0f : 1.0f;
        v16i af;
        constexpr int koff[8] = {0, 4, 8, 12, 32, 36, 40, 44};
#pragma unroll
        for (int j = 0; j < 16; ++j) {
            const int ko = ((j >> 3) << 6) + koff[j & 7] + (half << 4);
            const float4 f = *(const float4*)(xrow + dbase + ko);
            unsigned int bv;
            bv  = ((psgn * f.x > 0.0f) ? FP8_M1 : FP8_P1);
            bv |= ((psgn * f.y > 0.0f) ? FP8_M1 : FP8_P1) << 8;
            bv |= ((psgn * f.z > 0.0f) ? FP8_M1 : FP8_P1) << 16;
            bv |= ((psgn * f.w > 0.0f) ? FP8_M1 : FP8_P1) << 24;
            af[j] = (int)bv;
        }

#pragma unroll
        for (int c = 0; c < 16; ++c) {
            const int col = c * 16 + l16;
            const unsigned char* bp = ldsB + col * LDS_STRIDE + (half << 4);
            v16i bf;
#pragma unroll
            for (int g = 0; g < 4; ++g) {
                const int4 bq = *(const int4*)(bp + g * 32);
                bf[g * 4 + 0] = bq.x; bf[g * 4 + 1] = bq.y;
                bf[g * 4 + 2] = bq.z; bf[g * 4 + 3] = bq.w;
            }
            acc[c] = __builtin_amdgcn_wmma_f32_16x16x128_fp8_fp8(
                af, bf, (short)0, acc[c], false, false);
        }
    }

    unsigned int best[8];
#pragma unroll
    for (int r = 0; r < 8; ++r) best[r] = 0xFFFFFFFFu;

#pragma unroll
    for (int c = 0; c < 16; ++c) {
#pragma unroll
        for (int r = 0; r < 8; ++r) {
            const float dot    = acc[c][r];
            const float dist2f = (float)K_TOT - dot;
            const int   row    = rowstrip + r + half * 8;
            const int   col    = c * 16 + l16;
            dist_out[(size_t)row * T_TILES + col] = 0.5f * dist2f;
            const unsigned int key = (((unsigned int)dist2f) << 8) | (unsigned int)col;
            if (key < best[r]) best[r] = key;
        }
    }

#pragma unroll
    for (int r = 0; r < 8; ++r) {
        unsigned int k = best[r];
        unsigned int o;
        o = (unsigned int)__shfl_xor((int)k, 1, 32); k = (o < k) ? o : k;
        o = (unsigned int)__shfl_xor((int)k, 2, 32); k = (o < k) ? o : k;
        o = (unsigned int)__shfl_xor((int)k, 4, 32); k = (o < k) ? o : k;
        o = (unsigned int)__shfl_xor((int)k, 8, 32); k = (o < k) ? o : k;
        if (l16 == 0) idx_out[rowstrip + r + half * 8] = (int)(k & 0xFFu);
    }
}

extern "C" void kernel_launch(void* const* d_in, const int* in_sizes, int n_in,
                              void* d_out, int out_size, void* d_ws, size_t ws_size,
                              hipStream_t stream) {
    const float* x      = (const float*)d_in[0];
    const float* base   = (const float*)d_in[1];
    const float* deltas = (const float*)d_in[2];

    int*   idx_out  = (int*)d_out;
    float* dist_out = (float*)d_out + N_TOK;

    unsigned char* sigB = (unsigned char*)d_ws;

    build_sigs_kernel<<<(T_TILES * DIM_SZ) / 256, 256, 0, stream>>>(base, deltas, sigB);
    hamming_gemm_kernel<<<N_TOK / 128, 256, 0, stream>>>(x, sigB, idx_out, dist_out);
}
